// nativeContrastLoss_class_24876450578882
// MI455X (gfx1250) — matrix operand order repaired and checked by reading, not compiled, not run on hardware
//
#include <hip/hip_runtime.h>
#include <hip/hip_bf16.h>

// Supervised pixel-contrastive loss on MI455X (gfx1250, wave32).
// 16x16 WMMA tiles of the 8192x8192 logits matrix, never materialized.
// fp32 accuracy from f16 matrix cores via split: x = hi + lo (f16),
// x.y ~= hi.hi + hi.lo + lo.hi, each term on v_wmma_f32_16x16x32_f16.
// Three sweeps: (A) rowmax, (B) neg exp-sum + pos count/logit-sum,
// (C) sum over positives of log(exp(l-m) + neg_sum).
// This revision: 3 independent WMMA chains (no hazard v_nops) and a
// double-buffered, register-staged LDS pipeline (1 barrier/tile, global
// loads overlapped with WMMA compute).

#define M_ROWS 8192
#define DFEAT  64
#define WAVES  8
#define TPB    (WAVES * 32)            // 256 threads = 8 wave32
#define JSPLIT 8
#define NTILE  (M_ROWS / 16)           // 512 j-tiles
#define JT_PER (NTILE / JSPLIT)        // 64 tiles per j-chunk
#define NIB    (M_ROWS / (16 * WAVES)) // 64 i-blocks (128 rows each)
#define INV_T  10.0f                   // 1 / temperature

typedef __attribute__((ext_vector_type(16))) _Float16 v16h;
typedef __attribute__((ext_vector_type(8)))  _Float16 v8h;
typedef __attribute__((ext_vector_type(8)))  float    v8f;

union Frag { v16h v; v8h h[2]; };

// workspace layout (byte offsets); total use ~2.9 MB
#define XHI_OFF 0u
#define XLO_OFF (1u << 20)
#define P0_OFF  (2u << 20)
#define P1_OFF  ((2u << 20) + (256u << 10))
#define P2_OFF  ((2u << 20) + (512u << 10))
#define FM_OFF  ((2u << 20) + (768u << 10))
#define FN_OFF  (FM_OFF + (32u << 10))
#define FC_OFF  (FN_OFF + (32u << 10))
#define FP_OFF  (FC_OFF + (32u << 10))

enum { MODE_MAX = 0, MODE_SUM = 1, MODE_LOG = 2 };

// ---------------------------------------------------------------------------
// fp32 -> (hi, lo) f16 split
__global__ __launch_bounds__(256)
void prep_kernel(const float* __restrict__ X,
                 _Float16* __restrict__ Xhi, _Float16* __restrict__ Xlo) {
  int i = blockIdx.x * 256 + threadIdx.x;
  float v = X[i];
  _Float16 h = (_Float16)v;
  Xhi[i] = h;
  Xlo[i] = (_Float16)(v - (float)h);
}

// ---------------------------------------------------------------------------
// Tiled WMMA sweep. Each wave owns a 16-row i-tile; the 8 waves share
// double-buffered LDS-staged j-tiles. MODE selects the statistic produced.
template <int MODE>
__global__ __launch_bounds__(TPB)
void sweep_kernel(const _Float16* __restrict__ Xhi,
                  const _Float16* __restrict__ Xlo,
                  const int* __restrict__ labels,
                  const float* __restrict__ Fm,   // rowmax  (MODE_SUM/LOG)
                  const float* __restrict__ Fn,   // neg_sum (MODE_LOG)
                  float* __restrict__ P0, float* __restrict__ P1,
                  float* __restrict__ P2) {
  __shared__ __attribute__((aligned(16))) _Float16 sHi[2][16 * DFEAT];
  __shared__ __attribute__((aligned(16))) _Float16 sLo[2][16 * DFEAT];

  const int tid  = threadIdx.x;
  const int lane = tid & 31;
  const int wav  = tid >> 5;
  const int ln16 = lane & 15;
  const int hi   = lane >> 4;
  const int iw0  = blockIdx.x * (16 * WAVES) + wav * 16;
  const int jc   = blockIdx.y;

  // Resident A fragments (ISA 16-bit A layout: M = lane%16; per-lane halves
  // cover K = 32t + 8*hi + {0..7} and K = 32t + 16 + 8*hi + {0..7}).
  Frag aHi[2], aLo[2];
  {
    const _Float16* rHi = Xhi + (size_t)(iw0 + ln16) * DFEAT;
    const _Float16* rLo = Xlo + (size_t)(iw0 + ln16) * DFEAT;
#pragma unroll
    for (int t = 0; t < 2; ++t) {
      aHi[t].h[0] = *(const v8h*)(rHi + 32 * t + 8 * hi);
      aHi[t].h[1] = *(const v8h*)(rHi + 32 * t + 16 + 8 * hi);
      if (MODE != MODE_MAX) {
        aLo[t].h[0] = *(const v8h*)(rLo + 32 * t + 8 * hi);
        aLo[t].h[1] = *(const v8h*)(rLo + 32 * t + 16 + 8 * hi);
      }
    }
  }

  // Per-lane row metadata. C/D layout: accumulator element v belongs to
  // tile row (v + 8*hi), column lane%16.
  int rowg[8]; int li[8]; float mr[8]; float nr[8];
#pragma unroll
  for (int v = 0; v < 8; ++v) {
    rowg[v] = iw0 + v + 8 * hi;
    li[v] = 0; mr[v] = 0.f; nr[v] = 0.f;
    if (MODE != MODE_MAX) { li[v] = labels[rowg[v]]; mr[v] = Fm[rowg[v]]; }
    if (MODE == MODE_LOG) nr[v] = Fn[rowg[v]];
  }

  float a0[8], a1[8], a2[8];
#pragma unroll
  for (int v = 0; v < 8; ++v) {
    a0[v] = (MODE == MODE_MAX) ? -3.0e38f : 0.f;
    a1[v] = 0.f; a2[v] = 0.f;
  }

  const int jtBeg = jc * JT_PER;
  const int jtEnd = jtBeg + JT_PER;

  // Staging: thread tid owns one 16B chunk per tile.
  //   tid <  128        -> hi chunk tid   (row tid/8, k 8*(tid%8))
  //   tid in [128,256)  -> lo chunk tid-128 (only when MODE != MODE_MAX)
  const int cRow = (tid < 128) ? (tid >> 3) : ((tid - 128) >> 3);
  const int cKof = (tid < 128) ? 8 * (tid & 7) : 8 * ((tid - 128) & 7);
  const bool doHi = (tid < 128);
  const bool doLo = (tid >= 128) && (MODE != MODE_MAX);

  // prologue: stage first tile into buffer 0
  {
    const int j0 = jtBeg * 16;
    if (doHi)
      ((v8h*)sHi[0])[tid] =
          *(const v8h*)(Xhi + (size_t)(j0 + cRow) * DFEAT + cKof);
    else if (doLo)
      ((v8h*)sLo[0])[tid - 128] =
          *(const v8h*)(Xlo + (size_t)(j0 + cRow) * DFEAT + cKof);
  }

  int p = 0;
  for (int jt = jtBeg; jt < jtEnd; ++jt) {
    const int j0 = jt * 16;
    __syncthreads();  // buf[p] staged; all reads of buf[p^1] finished

    // kick off global loads for the next tile (latency hidden by WMMA below)
    const bool hasNext = (jt + 1 < jtEnd);
    v8h nxt = {};
    if (hasNext) {
      const int n0 = j0 + 16;
      if (doHi)
        nxt = *(const v8h*)(Xhi + (size_t)(n0 + cRow) * DFEAT + cKof);
      else if (doLo)
        nxt = *(const v8h*)(Xlo + (size_t)(n0 + cRow) * DFEAT + cKof);
    }
    const int lj = (MODE != MODE_MAX) ? labels[j0 + ln16] : 0;

    // B fragments (ISA B layout: N = lane%16; halves cover
    // K = 32t + 8*hi + {0..7} and K = 32t + 16 + 8*hi + {0..7}).
    Frag bHi[2], bLo[2];
    const _Float16* cHi = sHi[p] + ln16 * DFEAT;
    const _Float16* cLo = sLo[p] + ln16 * DFEAT;
#pragma unroll
    for (int t = 0; t < 2; ++t) {
      bHi[t].h[0] = *(const v8h*)(cHi + 32 * t + 8 * hi);
      bHi[t].h[1] = *(const v8h*)(cHi + 32 * t + 16 + 8 * hi);
      if (MODE != MODE_MAX) {
        bLo[t].h[0] = *(const v8h*)(cLo + 32 * t + 8 * hi);
        bLo[t].h[1] = *(const v8h*)(cLo + 32 * t + 16 + 8 * hi);
      }
    }

    // Independent WMMA chains (no D->C serialization between chains).
    v8f c0 = {}, c1 = {}, c2 = {};
    c0 = __builtin_amdgcn_wmma_f32_16x16x32_f16(false, aHi[0].v, false,
                                                bHi[0].v, (short)0, c0,
                                                false, false);
    if (MODE != MODE_MAX) {
      c1 = __builtin_amdgcn_wmma_f32_16x16x32_f16(false, aHi[0].v, false,
                                                  bLo[0].v, (short)0, c1,
                                                  false, false);
      c2 = __builtin_amdgcn_wmma_f32_16x16x32_f16(false, aLo[0].v, false,
                                                  bHi[0].v, (short)0, c2,
                                                  false, false);
    }
    c0 = __builtin_amdgcn_wmma_f32_16x16x32_f16(false, aHi[1].v, false,
                                                bHi[1].v, (short)0, c0,
                                                false, false);
    if (MODE != MODE_MAX) {
      c1 = __builtin_amdgcn_wmma_f32_16x16x32_f16(false, aHi[1].v, false,
                                                  bLo[1].v, (short)0, c1,
                                                  false, false);
      c2 = __builtin_amdgcn_wmma_f32_16x16x32_f16(false, aLo[1].v, false,
                                                  bHi[1].v, (short)0, c2,
                                                  false, false);
    }

    const int colg = j0 + ln16;
#pragma unroll
    for (int v = 0; v < 8; ++v) {
      float l = (MODE == MODE_MAX) ? c0[v] * INV_T
                                   : (c0[v] + c1[v] + c2[v]) * INV_T;
      if (MODE == MODE_MAX) {
        a0[v] = fmaxf(a0[v], l);
      } else if (MODE == MODE_SUM) {
        bool same = (lj == li[v]);
        if (!same) {
          a0[v] += __expf(l - mr[v]);             // negative exp-sum
        } else if (colg != rowg[v]) {
          a1[v] += 1.0f;                           // positive count
          a2[v] += (l - mr[v]);                    // positive shifted-logit sum
        }
      } else {  // MODE_LOG
        if (lj == li[v] && colg != rowg[v])
          a0[v] += __logf(__expf(l - mr[v]) + nr[v]);
      }
    }

    // commit the staged next tile into the other buffer
    if (hasNext) {
      if (doHi)      ((v8h*)sHi[p ^ 1])[tid] = nxt;
      else if (doLo) ((v8h*)sLo[p ^ 1])[tid - 128] = nxt;
    }
    p ^= 1;
  }

  // merge across the 16 lanes of each half (row = v + 8*hi, all 16 columns)
#pragma unroll
  for (int s = 1; s < 16; s <<= 1) {
#pragma unroll
    for (int v = 0; v < 8; ++v) {
      float o0 = __shfl_xor(a0[v], s, 32);
      a0[v] = (MODE == MODE_MAX) ? fmaxf(a0[v], o0) : (a0[v] + o0);
      if (MODE == MODE_SUM) {
        a1[v] += __shfl_xor(a1[v], s, 32);
        a2[v] += __shfl_xor(a2[v], s, 32);
      }
    }
  }
  if (ln16 == 0) {
#pragma unroll
    for (int v = 0; v < 8; ++v) {
      size_t idx = (size_t)jc * M_ROWS + rowg[v];
      P0[idx] = a0[v];
      if (MODE == MODE_SUM) { P1[idx] = a1[v]; P2[idx] = a2[v]; }
    }
  }
}

// ---------------------------------------------------------------------------
__global__ __launch_bounds__(256)
void reduce_max_kernel(const float* __restrict__ P0, float* __restrict__ Fm) {
  int r = blockIdx.x * 256 + threadIdx.x;
  float m = -3.0e38f;
  for (int c = 0; c < JSPLIT; ++c) m = fmaxf(m, P0[(size_t)c * M_ROWS + r]);
  Fm[r] = m;
}

__global__ __launch_bounds__(256)
void reduce_sum3_kernel(const float* __restrict__ P0, const float* __restrict__ P1,
                        const float* __restrict__ P2, float* __restrict__ Fn,
                        float* __restrict__ Fc, float* __restrict__ Fp) {
  int r = blockIdx.x * 256 + threadIdx.x;
  float s0 = 0.f, s1 = 0.f, s2 = 0.f;
  for (int c = 0; c < JSPLIT; ++c) {
    size_t idx = (size_t)c * M_ROWS + r;
    s0 += P0[idx]; s1 += P1[idx]; s2 += P2[idx];
  }
  Fn[r] = s0; Fc[r] = s1; Fp[r] = s2;
}

__global__ __launch_bounds__(256)
void finalize_kernel(const float* __restrict__ Pd, const float* __restrict__ Fp,
                     const float* __restrict__ Fc, float* __restrict__ out) {
  __shared__ float red[256];
  float s = 0.f;
  for (int r = threadIdx.x; r < M_ROWS; r += 256) {
    float ds = 0.f;
    for (int c = 0; c < JSPLIT; ++c) ds += Pd[(size_t)c * M_ROWS + r];
    s += (Fp[r] - ds) / Fc[r];   // mean_log_prob_pos
  }
  red[threadIdx.x] = s;
  __syncthreads();
  for (int k = 128; k > 0; k >>= 1) {
    if ((int)threadIdx.x < k) red[threadIdx.x] += red[threadIdx.x + k];
    __syncthreads();
  }
  if (threadIdx.x == 0)
    out[0] = -(0.1f / 1.0f) * red[0] / (float)M_ROWS;  // -(T/baseT) * mean
}

// ---------------------------------------------------------------------------
extern "C" void kernel_launch(void* const* d_in, const int* in_sizes, int n_in,
                              void* d_out, int out_size, void* d_ws, size_t ws_size,
                              hipStream_t stream) {
  const float* X = (const float*)d_in[0];
  const int* y   = (const int*)d_in[1];
  char* ws = (char*)d_ws;
  _Float16* Xhi = (_Float16*)(ws + XHI_OFF);
  _Float16* Xlo = (_Float16*)(ws + XLO_OFF);
  float* P0 = (float*)(ws + P0_OFF);
  float* P1 = (float*)(ws + P1_OFF);
  float* P2 = (float*)(ws + P2_OFF);
  float* Fm = (float*)(ws + FM_OFF);
  float* Fn = (float*)(ws + FN_OFF);
  float* Fc = (float*)(ws + FC_OFF);
  float* Fp = (float*)(ws + FP_OFF);

  prep_kernel<<<(M_ROWS * DFEAT) / 256, 256, 0, stream>>>(X, Xhi, Xlo);

  dim3 g(NIB, JSPLIT);
  sweep_kernel<MODE_MAX><<<g, TPB, 0, stream>>>(Xhi, Xlo, y, Fm, Fn, P0, P1, P2);
  reduce_max_kernel<<<M_ROWS / 256, 256, 0, stream>>>(P0, Fm);

  sweep_kernel<MODE_SUM><<<g, TPB, 0, stream>>>(Xhi, Xlo, y, Fm, Fn, P0, P1, P2);
  reduce_sum3_kernel<<<M_ROWS / 256, 256, 0, stream>>>(P0, P1, P2, Fn, Fc, Fp);

  sweep_kernel<MODE_LOG><<<g, TPB, 0, stream>>>(Xhi, Xlo, y, Fm, Fn, P0, P1, P2);
  finalize_kernel<<<1, 256, 0, stream>>>(P0, Fp, Fc, (float*)d_out);
}
